// ODEModel_67250597920774
// MI455X (gfx1250) — matrix operand order repaired and checked by reading, not compiled, not run on hardware
//
#include <hip/hip_runtime.h>
#include <stdint.h>

// ---- problem constants (from reference) ----
#define B_SZ   1024
#define IN_SZ  256
#define LAT_SZ 512
#define HID_SZ 1024
#define OUT_SZ 64
#define NSTEPS 20

typedef __attribute__((ext_vector_type(16))) __bf16 v16bf;
typedef __attribute__((ext_vector_type(8)))  float  v8f;

union Frag16 { v16bf v; uint32_t u[8]; };

// Packed f32x2 -> bf16x2 convert (RNE) in a single VALU op.
// V_CVT_PK_BF16_F32: D[15:0] = bf16(S0), D[31:16] = bf16(S1).
__device__ __forceinline__ uint32_t pack_bf(float lo, float hi) {
  uint32_t r;
  asm("v_cvt_pk_bf16_f32 %0, %1, %2" : "=v"(r) : "v"(lo), "v"(hi));
  return r;
}
__device__ __forceinline__ uint16_t f2bf(float f) {
  union { __bf16 h; uint16_t u; } c;
  c.h = (__bf16)f;
  return c.u;
}

// Transpose + convert: src fp32 [R][C] row-major -> dst bf16 [C][R] row-major
__global__ void convT_kernel(const float* __restrict__ src,
                             uint16_t* __restrict__ dst, int R, int C) {
  int idx = blockIdx.x * 256 + threadIdx.x;
  if (idx >= R * C) return;
  int r = idx / C, c = idx % C;
  dst[(size_t)c * R + r] = f2bf(src[idx]);
}

// bias_t[j] = b1[j] + t * W1[LAT, j]   (folds the time column of z into bias)
__global__ void biasrow_kernel(const float* __restrict__ b1,
                               const float* __restrict__ w1last,
                               float t, float* __restrict__ out) {
  int i = blockIdx.x * 256 + threadIdx.x;
  if (i < HID_SZ) out[i] = b1[i] + t * w1last[i];
}

// Generic bf16-WMMA GEMM, M = 1024 fixed via grid, K multiple of 32.
//   A (fp32) = A1 + coef*A2 (A2 may be null), leading dim lda
//   BT (bf16) stored [N][K] row-major (pre-transposed weights)
//   MODE 0: Out = tanh(A@B + bias)
//   MODE 1: r = A@B + bias; Out = r; Acc = (AccInit? AccInit : Acc) + accw*r
//   MODE 2: Out = A@B + bias
// Software-pipelined: double-buffered LDS; global fetch of chunk k+1 overlaps
// WMMA compute of chunk k; two barriers per chunk.
template <int MODE>
__global__ __launch_bounds__(256) void gemm_wmma_kernel(
    const float* __restrict__ A1, const float* __restrict__ A2, float coef, int lda,
    const uint16_t* __restrict__ BT, const float* __restrict__ bias,
    float* __restrict__ Out, int N, int K,
    float accw, float* __restrict__ Acc, const float* __restrict__ AccInit)
{
  __shared__ uint32_t lA[2][128 * 17];   // 128 rows x 32 bf16 (16 dwords, pad 17)
  __shared__ uint32_t lB[2][128 * 17];   // 128 N-rows x 32 bf16

  const int tid  = threadIdx.x;
  const int lane = tid & 31;
  const int wave = tid >> 5;
  const int wm = wave >> 1;                 // 0..3 -> 32 rows each
  const int wn = wave & 1;                  // 0..1 -> 64 cols each
  const int rowBlock = blockIdx.y * 128;
  const int colBlock = blockIdx.x * 128;
  const int sr   = tid >> 1;                // staging row 0..127
  const int half = tid & 1;                 // staging K-half (16 elems)

  // per-wave LDS fragment base offsets (dwords); +mi*16*17 / +ni*16*17 per tile
  const int aoff = (wm * 32 + (lane & 15)) * 17 + ((lane < 16) ? 0 : 4);
  const int boff = (wn * 64 + (lane & 15)) * 17 + ((lane < 16) ? 0 : 4);

  v8f acc[2][4] = {};                       // fp32 accumulators

  // ---- staging fetch helpers (global -> registers) ----
  float    za[16];
  uint32_t zb[8];
  auto fetchA = [&](int k0) {
    const size_t off = (size_t)(rowBlock + sr) * lda + (k0 + half * 16);
    const float* s1 = A1 + off;
    if (A2) {
      const float* s2 = A2 + off;
      #pragma unroll
      for (int i = 0; i < 16; ++i) za[i] = s1[i] + coef * s2[i];
    } else {
      #pragma unroll
      for (int i = 0; i < 16; ++i) za[i] = s1[i];
    }
  };
  auto fetchB = [&](int k0) {
    const int gn = colBlock + sr;
    if (gn < N) {
      const uint32_t* s = (const uint32_t*)(BT + (size_t)gn * K + (k0 + half * 16));
      #pragma unroll
      for (int i = 0; i < 8; ++i) zb[i] = s[i];
    } else {
      #pragma unroll
      for (int i = 0; i < 8; ++i) zb[i] = 0u;
    }
  };

  const int nk = K >> 5;
  fetchA(0);
  fetchB(0);

  for (int kt = 0; kt < nk; ++kt) {
    const int buf = kt & 1;
    // ---- commit staged registers to LDS buffer `buf` ----
    {
      uint32_t* da = &lA[buf][sr * 17 + half * 8];
      #pragma unroll
      for (int i = 0; i < 8; ++i) da[i] = pack_bf(za[2 * i], za[2 * i + 1]);
      uint32_t* db = &lB[buf][sr * 17 + half * 8];
      #pragma unroll
      for (int i = 0; i < 8; ++i) db[i] = zb[i];
    }
    __syncthreads();

    // ---- prefetch next chunk into registers (overlaps with WMMA below) ----
    if (kt + 1 < nk) {
      fetchA((kt + 1) << 5);
      fetchB((kt + 1) << 5);
    }

    // ---- load ALL fragments first, then burst 8 independent WMMAs ----
    const uint32_t* LA = lA[buf];
    const uint32_t* LB = lB[buf];
    Frag16 af[2];
    #pragma unroll
    for (int mi = 0; mi < 2; ++mi) {
      const uint32_t* base = &LA[aoff + mi * 16 * 17];
      #pragma unroll
      for (int i = 0; i < 4; ++i) { af[mi].u[i] = base[i]; af[mi].u[4 + i] = base[8 + i]; }
    }
    Frag16 bfv[4];
    #pragma unroll
    for (int ni = 0; ni < 4; ++ni) {
      const uint32_t* base = &LB[boff + ni * 16 * 17];
      #pragma unroll
      for (int i = 0; i < 4; ++i) { bfv[ni].u[i] = base[i]; bfv[ni].u[4 + i] = base[8 + i]; }
    }
    #pragma unroll
    for (int ni = 0; ni < 4; ++ni) {
      #pragma unroll
      for (int mi = 0; mi < 2; ++mi) {
        acc[mi][ni] = __builtin_amdgcn_wmma_f32_16x16x32_bf16(
            false, af[mi].v, false, bfv[ni].v, (short)0, acc[mi][ni], false, false);
      }
    }
    __syncthreads();
  }

  // ---- epilogue: VGPR v -> row v + (lane>=16 ? 8 : 0), col = lane%16 ----
  #pragma unroll
  for (int ni = 0; ni < 4; ++ni) {
    const int col = colBlock + wn * 64 + ni * 16 + (lane & 15);
    if (col >= N) continue;
    const float bv = bias[col];
    const int rbase = rowBlock + wm * 32 + ((lane < 16) ? 0 : 8);
    #pragma unroll
    for (int mi = 0; mi < 2; ++mi) {
      #pragma unroll
      for (int v = 0; v < 8; ++v) {
        const int row = rbase + mi * 16 + v;
        const float r = acc[mi][ni][v] + bv;
        const size_t o = (size_t)row * N + col;
        if (MODE == 0) {
          Out[o] = tanhf(r);
        } else if (MODE == 1) {
          Out[o] = r;
          const float prev = AccInit ? AccInit[o] : Acc[o];
          Acc[o] = prev + accw * r;
        } else {
          Out[o] = r;
        }
      }
    }
  }
}

extern "C" void kernel_launch(void* const* d_in, const int* in_sizes, int n_in,
                              void* d_out, int out_size, void* d_ws, size_t ws_size,
                              hipStream_t stream) {
  const float* x     = (const float*)d_in[0];
  const float* W_in  = (const float*)d_in[1];
  const float* b_in  = (const float*)d_in[2];
  const float* W1    = (const float*)d_in[3];
  const float* b1    = (const float*)d_in[4];
  const float* W2    = (const float*)d_in[5];
  const float* b2    = (const float*)d_in[6];
  const float* W_out = (const float*)d_in[7];
  const float* b_out = (const float*)d_in[8];
  float* out = (float*)d_out;

  // deterministic bump allocator over d_ws
  char* wsp = (char*)d_ws;
  auto alloc = [&](size_t bytes) -> char* {
    char* p = wsp;
    wsp += (bytes + 255) & ~(size_t)255;
    return p;
  };
  uint16_t* WinT  = (uint16_t*)alloc((size_t)LAT_SZ * IN_SZ * 2);          // [LAT][IN]
  uint16_t* W1T   = (uint16_t*)alloc((size_t)HID_SZ * LAT_SZ * 2);         // [HID][LAT]
  uint16_t* W2T   = (uint16_t*)alloc((size_t)LAT_SZ * HID_SZ * 2);         // [LAT][HID]
  uint16_t* WoutT = (uint16_t*)alloc((size_t)OUT_SZ * (LAT_SZ / 2) * 2);   // [OUT][LAT/2]
  float* biasbuf  = (float*)alloc((size_t)HID_SZ * 4);
  float* hbuf0    = (float*)alloc((size_t)B_SZ * LAT_SZ * 4);
  float* hbuf1    = (float*)alloc((size_t)B_SZ * LAT_SZ * 4);
  float* kbuf     = (float*)alloc((size_t)B_SZ * LAT_SZ * 4);
  float* Htmp     = (float*)alloc((size_t)B_SZ * HID_SZ * 4);

  // weight conversion + transpose to bf16 [N][K]
  convT_kernel<<<(IN_SZ * LAT_SZ + 255) / 256, 256, 0, stream>>>(W_in, WinT, IN_SZ, LAT_SZ);
  convT_kernel<<<(LAT_SZ * HID_SZ + 255) / 256, 256, 0, stream>>>(W1, W1T, LAT_SZ, HID_SZ);
  convT_kernel<<<(HID_SZ * LAT_SZ + 255) / 256, 256, 0, stream>>>(W2, W2T, HID_SZ, LAT_SZ);
  convT_kernel<<<((LAT_SZ / 2) * OUT_SZ + 255) / 256, 256, 0, stream>>>(W_out, WoutT, LAT_SZ / 2, OUT_SZ);

  const float dt = 1.0f / NSTEPS;

  // h0 = tanh(x @ W_in + b_in)
  {
    dim3 g(LAT_SZ / 128, B_SZ / 128);
    gemm_wmma_kernel<0><<<g, 256, 0, stream>>>(x, nullptr, 0.0f, IN_SZ, WinT, b_in,
                                               hbuf0, LAT_SZ, IN_SZ, 0.0f, nullptr, nullptr);
  }

  float* h   = hbuf0;
  float* acc = hbuf1;
  const float cst[4] = {0.0f, 0.5f * dt, 0.5f * dt, dt};            // z = h + c*k_prev
  const float wst[4] = {dt / 6.0f, dt / 3.0f, dt / 3.0f, dt / 6.0f};
  const float tst[4] = {0.0f, 0.5f * dt, 0.5f * dt, dt};            // stage time offsets

  for (int s = 0; s < NSTEPS; ++s) {
    const float t0 = dt * (float)s;
    for (int st = 0; st < 4; ++st) {
      biasrow_kernel<<<HID_SZ / 256, 256, 0, stream>>>(
          b1, W1 + (size_t)LAT_SZ * HID_SZ, t0 + tst[st], biasbuf);
      // Htmp = tanh((h + c*k_prev) @ W1[:512] + bias_t)
      {
        dim3 g(HID_SZ / 128, B_SZ / 128);
        gemm_wmma_kernel<0><<<g, 256, 0, stream>>>(
            h, (st == 0) ? nullptr : kbuf, cst[st], LAT_SZ, W1T, biasbuf,
            Htmp, HID_SZ, LAT_SZ, 0.0f, nullptr, nullptr);
      }
      // k_st = Htmp @ W2 + b2 ; acc = (st==0 ? h : acc) + w_st * k_st
      {
        dim3 g(LAT_SZ / 128, B_SZ / 128);
        gemm_wmma_kernel<1><<<g, 256, 0, stream>>>(
            Htmp, nullptr, 0.0f, HID_SZ, W2T, b2,
            kbuf, LAT_SZ, HID_SZ, wst[st], acc, (st == 0) ? h : nullptr);
      }
    }
    float* tmp = h; h = acc; acc = tmp;   // h_{s+1} = acc
  }

  // out = h[:, :256] @ W_out + b_out
  {
    dim3 g(1, B_SZ / 128);
    gemm_wmma_kernel<2><<<g, 256, 0, stream>>>(h, nullptr, 0.0f, LAT_SZ, WoutT, b_out,
                                               out, OUT_SZ, LAT_SZ / 2, 0.0f, nullptr, nullptr);
  }
}
